// GAT_layer_27376121545350
// MI455X (gfx1250) — matrix operand order repaired and checked by reading, not compiled, not run on hardware
//
#include <hip/hip_runtime.h>
#include <hip/hip_bf16.h>

// ---------------------------------------------------------------------------
// GAT layer for MI455X (gfx1250, wave32, WMMA bf16).
//   N=100000 nodes, E=1600000 edges, IN=128, EM=64, HI=64, OUT=128, H=4 heads
// ---------------------------------------------------------------------------

#define NNODES 100000
#define NEDGES 1600000
#define INDIM  128
#define EMDIM  64
#define HIDIM  64
#define OUTDIM 128
#define HEADS  4

typedef __attribute__((ext_vector_type(16))) __bf16 v16bf;
typedef __attribute__((ext_vector_type(8)))  __bf16 v8bf;
typedef __attribute__((ext_vector_type(2)))  __bf16 v2bf;
typedef __attribute__((ext_vector_type(8)))  float  v8f;

__device__ __forceinline__ float leaky(float v) {
    return v > 0.0f ? v : 0.01f * v;
}

__device__ __forceinline__ v8f wmma_bf16(v16bf a, v16bf b, v8f c) {
    return __builtin_amdgcn_wmma_f32_16x16x32_bf16(
        false, a, false, b, (short)0, c, false, false);
}

// ---- A fragment from a bf16 LDS tile (CDNA5 16-bit A 16x32 layout) --------
// lane -> M = lane&15, half = lane>>4; element e -> K = kbase+(e>>3)*16+half*8+(e&7)
// => two contiguous 8-element runs at (kbase+half*8) and (kbase+16+half*8).
__device__ __forceinline__ v16bf fragA_lds(const __bf16* tile, int lda,
                                           int m, int half, int kbase) {
    const __bf16* p = tile + m * lda + kbase + half * 8;
    v8bf lo = *(const v8bf*)p;          // ds_load_b128
    v8bf hi = *(const v8bf*)(p + 16);   // ds_load_b128
    return __builtin_shufflevector(lo, hi, 0,1,2,3,4,5,6,7,8,9,10,11,12,13,14,15);
}

// ---- B fragment from pre-packed bf16 weights ------------------------------
// packed[((frag*32 + lane)*16 + e] holds B element K=kbase+(e>>3)*16+half*8+(e&7), N=nt*16+(lane&15)
__device__ __forceinline__ v16bf fragB_packed(const __bf16* packed, int frag, int lane) {
    const __bf16* p = packed + ((size_t)frag * 32 + lane) * 16;
    v8bf lo = *(const v8bf*)p;          // global_load_b128
    v8bf hi = *(const v8bf*)(p + 8);    // global_load_b128
    return __builtin_shufflevector(lo, hi, 0,1,2,3,4,5,6,7,8,9,10,11,12,13,14,15);
}

// ---- C fragment writeout to bf16 LDS tile ---------------------------------
__device__ __forceinline__ void storeC_lds(__bf16* tile, int lda, int n, int half,
                                           v8f acc, float bias, bool act) {
#pragma unroll
    for (int r = 0; r < 8; ++r) {
        float v = acc[r] + bias;
        if (act) v = leaky(v);
        tile[(r + 8 * half) * lda + n] = (__bf16)v;
    }
}

// ---------------------------------------------------------------------------
// Weight repack: f32 [batch][K(src rows)][N] -> bf16 fragment-ready layout.
// dst flat index = ((b*NT*KS + nt*KS + kt)*32 + lane)*16 + e
//   n = nt*16 + (lane&15);  k = kt*32 + (e>>3)*16 + (lane>>4)*8 + (e&7)  (+rowOff in source)
// ---------------------------------------------------------------------------
__global__ __launch_bounds__(256) void pack_w_kernel(
    const float* __restrict__ src, __bf16* __restrict__ dst,
    int K, int N, int batch, int rowOff, size_t batchStride)
{
    int idx = blockIdx.x * 256 + threadIdx.x;
    int total = batch * K * N;
    if (idx >= total) return;
    int e    = idx & 15;
    int lane = (idx >> 4) & 31;
    int frag = idx >> 9;
    int KS = K >> 5;
    int NT = N >> 4;
    int kt = frag % KS;
    int nt = (frag / KS) % NT;
    int b  = frag / (KS * NT);
    int n = nt * 16 + (lane & 15);
    int k = kt * 32 + (e >> 3) * 16 + (lane >> 4) * 8 + (e & 7);
    dst[idx] = (__bf16)src[(size_t)b * batchStride + (size_t)(rowOff + k) * N + n];
}

// ---------------------------------------------------------------------------
// Kernel 1: per-head node embedding MLP + attention projections.
//   hs   = (leaky(x@Wi1+bi1))@Wi2 + bi2     -> bf16 [H,N,64]
//   ssrc = hs @ Wa1[:, :64]                 -> bf16 [H,N,64]
//   sdst = hs @ Wa1[:, 64:]                 -> bf16 [H,N,64]
// One 128-thread block (4 waves) per 16-node tile; wave == head.
// ---------------------------------------------------------------------------
__global__ __launch_bounds__(128) void node_embed_kernel(
    const float* __restrict__ x,
    const __bf16* __restrict__ Wi1p, const float* __restrict__ bi1,
    const __bf16* __restrict__ Wi2p, const float* __restrict__ bi2,
    const __bf16* __restrict__ Wa1ap, const __bf16* __restrict__ Wa1bp,
    __bf16* __restrict__ hs_out,
    __bf16* __restrict__ ssrc_out,
    __bf16* __restrict__ sdst_out)
{
    __shared__ __bf16 xs[16 * INDIM];           // 4 KB
    __shared__ __bf16 h1s[HEADS][16 * EMDIM];   // 8 KB
    __shared__ __bf16 hss[HEADS][16 * EMDIM];   // 8 KB

    const int nb  = blockIdx.x * 16;
    const int tid = threadIdx.x;

    // cooperative x-tile load: f32 pairs -> packed bf16 (cvt_pk), coalesced
#pragma unroll
    for (int i = 0; i < 8; ++i) {
        int pidx = i * 128 + tid;               // 1024 pairs
        int row = pidx >> 6;
        int cp  = pidx & 63;
        float2 v = *(const float2*)&x[(size_t)(nb + row) * INDIM + 2 * cp];
        v2bf b; b[0] = (__bf16)v.x; b[1] = (__bf16)v.y;
        *(v2bf*)&xs[row * INDIM + 2 * cp] = b;
    }
    __syncthreads();

    const int h = tid >> 5, lane = tid & 31;
    const int half = lane >> 4, m = lane & 15;

    // ---- GEMM1: h1 = leaky(x @ Wi1 + bi1)   (16x128x64) ----
    {
        v16bf a[4];
#pragma unroll
        for (int kt = 0; kt < 4; ++kt)
            a[kt] = fragA_lds(xs, INDIM, m, half, kt * 32);
#pragma unroll
        for (int nt = 0; nt < 4; ++nt) {
            v8f acc = {};
#pragma unroll
            for (int kt = 0; kt < 4; ++kt)
                acc = wmma_bf16(a[kt],
                                fragB_packed(Wi1p, (h * 4 + nt) * 4 + kt, lane), acc);
            int n = nt * 16 + m;
            storeC_lds(h1s[h], EMDIM, n, half, acc, bi1[h * EMDIM + n], true);
        }
    }

    // ---- GEMM2: hs = h1 @ Wi2 + bi2   (16x64x64) ----
    {
        v16bf a[2];
#pragma unroll
        for (int kt = 0; kt < 2; ++kt)
            a[kt] = fragA_lds(h1s[h], EMDIM, m, half, kt * 32);
#pragma unroll
        for (int nt = 0; nt < 4; ++nt) {
            v8f acc = {};
#pragma unroll
            for (int kt = 0; kt < 2; ++kt)
                acc = wmma_bf16(a[kt],
                                fragB_packed(Wi2p, (h * 4 + nt) * 2 + kt, lane), acc);
            int n = nt * 16 + m;
            storeC_lds(hss[h], EMDIM, n, half, acc, bi2[h * EMDIM + n], false);
        }
    }
    // dump hss -> hs_out (coalesced b128)
    {
        __bf16* g = hs_out + ((size_t)h * NNODES + nb) * EMDIM;
#pragma unroll
        for (int i = 0; i < 4; ++i) {
            int o = i * 256 + lane * 8;
            *(v8bf*)&g[o] = *(const v8bf*)&hss[h][o];
        }
    }

    // ---- GEMM3/4: ssrc = hs@Wa1[:, :64], sdst = hs@Wa1[:, 64:] (shared A) --
    {
        v16bf a[2];
#pragma unroll
        for (int kt = 0; kt < 2; ++kt)
            a[kt] = fragA_lds(hss[h], EMDIM, m, half, kt * 32);

        // ssrc
#pragma unroll
        for (int nt = 0; nt < 4; ++nt) {
            v8f acc = {};
#pragma unroll
            for (int kt = 0; kt < 2; ++kt)
                acc = wmma_bf16(a[kt],
                                fragB_packed(Wa1ap, (h * 4 + nt) * 2 + kt, lane), acc);
            storeC_lds(h1s[h], HIDIM, nt * 16 + m, half, acc, 0.0f, false);
        }
        {
            __bf16* g = ssrc_out + ((size_t)h * NNODES + nb) * HIDIM;
#pragma unroll
            for (int i = 0; i < 4; ++i) {
                int o = i * 256 + lane * 8;
                *(v8bf*)&g[o] = *(const v8bf*)&h1s[h][o];
            }
        }

        // sdst
#pragma unroll
        for (int nt = 0; nt < 4; ++nt) {
            v8f acc = {};
#pragma unroll
            for (int kt = 0; kt < 2; ++kt)
                acc = wmma_bf16(a[kt],
                                fragB_packed(Wa1bp, (h * 4 + nt) * 2 + kt, lane), acc);
            storeC_lds(h1s[h], HIDIM, nt * 16 + m, half, acc, 0.0f, false);
        }
        {
            __bf16* g = sdst_out + ((size_t)h * NNODES + nb) * HIDIM;
#pragma unroll
            for (int i = 0; i < 4; ++i) {
                int o = i * 256 + lane * 8;
                *(v8bf*)&g[o] = *(const v8bf*)&h1s[h][o];
            }
        }
    }
}

// ---------------------------------------------------------------------------
// Kernel 2: per-edge attention logits + exp + softmax-denominator scatter.
// One wave per edge; head h = lane>>3; 8 lanes/head, 8 k each (16B loads).
// ---------------------------------------------------------------------------
__global__ __launch_bounds__(256) void edge_attn_kernel(
    const int* __restrict__ ei,
    const __bf16* __restrict__ ssrc, const __bf16* __restrict__ sdst,
    const float* __restrict__ ba1, const float* __restrict__ Wa2,
    const float* __restrict__ ba2,
    float* __restrict__ e_out, float* __restrict__ esum)
{
    __shared__ float sba1[HEADS * HIDIM];
    __shared__ float swa2[HEADS * HIDIM];
    __shared__ float sba2[HEADS];
    const int tid = threadIdx.x;
    sba1[tid] = ba1[tid];
    swa2[tid] = Wa2[tid];
    if (tid < HEADS) sba2[tid] = ba2[tid];
    __syncthreads();

    const int wave = tid >> 5;
    const int lane = tid & 31;
    const long e = (long)blockIdx.x * 8 + wave;
    if (e >= NEDGES) return;

    const int src = ei[e];
    const int dst = ei[NEDGES + e];
    const int h  = lane >> 3;
    const int j  = lane & 7;
    const int k0 = j * 8;

    v8bf sv = *(const v8bf*)(ssrc + ((size_t)h * NNODES + src) * HIDIM + k0);
    v8bf dv = *(const v8bf*)(sdst + ((size_t)h * NNODES + dst) * HIDIM + k0);

    float part = 0.0f;
#pragma unroll
    for (int i = 0; i < 8; ++i) {
        float t = (float)sv[i] + (float)dv[i] + sba1[h * HIDIM + k0 + i];
        part += leaky(t) * swa2[h * HIDIM + k0 + i];
    }
    part += __shfl_xor(part, 1, 32);
    part += __shfl_xor(part, 2, 32);
    part += __shfl_xor(part, 4, 32);

    if (j == 0) {
        float a  = leaky(part + sba2[h]);
        float ev = __expf(a);
        e_out[(size_t)h * NEDGES + e] = ev;
        unsafeAtomicAdd(&esum[(size_t)h * NNODES + src], ev);
    }
}

// ---------------------------------------------------------------------------
// Kernel 3: alpha = e / esum[src]; scatter-add alpha * hs[src] onto agg[dst].
// ---------------------------------------------------------------------------
__global__ __launch_bounds__(256) void edge_aggr_kernel(
    const int* __restrict__ ei,
    const __bf16* __restrict__ hs,
    const float* __restrict__ e_in, const float* __restrict__ esum,
    float* __restrict__ agg)
{
    const int wave = threadIdx.x >> 5;
    const int lane = threadIdx.x & 31;
    const long e = (long)blockIdx.x * 8 + wave;
    if (e >= NEDGES) return;

    const int src = ei[e];
    const int dst = ei[NEDGES + e];
    const int h  = lane >> 3;
    const int j  = lane & 7;
    const int k0 = j * 8;

    float alpha = 0.0f;
    if (j == 0)
        alpha = e_in[(size_t)h * NEDGES + e] / esum[(size_t)h * NNODES + src];
    alpha = __shfl(alpha, h << 3, 32);

    v8bf hv = *(const v8bf*)(hs + ((size_t)h * NNODES + src) * EMDIM + k0);
    float* pa = agg + ((size_t)h * NNODES + dst) * EMDIM + k0;
#pragma unroll
    for (int i = 0; i < 8; ++i) {
        unsafeAtomicAdd(&pa[i], (float)hv[i] * alpha);
    }
}

// ---------------------------------------------------------------------------
// Kernel 4: out = leaky(hcat @ Wf1 + bf1) @ Wf2 + bf2.
// hcat[n][h*64+k] = agg[h][n][k].  One wave per 16-node tile.
// ---------------------------------------------------------------------------
__global__ __launch_bounds__(32) void final_mlp_kernel(
    const float* __restrict__ agg,
    const __bf16* __restrict__ Wf1p, const float* __restrict__ bf1,
    const __bf16* __restrict__ Wf2p, const float* __restrict__ bf2,
    float* __restrict__ out)
{
    __shared__ __bf16 hc[16 * (HEADS * EMDIM)];  // 16x256 bf16 = 8 KB
    __shared__ __bf16 mid[16 * HIDIM];           // 16x64  bf16 = 2 KB
    __shared__ float  outs[16 * OUTDIM];         // 16x128 f32  = 8 KB

    const int nb   = blockIdx.x * 16;
    const int lane = threadIdx.x;

    // fill hc: pairs of f32 from agg -> packed bf16
#pragma unroll
    for (int i = 0; i < 64; ++i) {
        int pidx = i * 32 + lane;                // 2048 pairs
        int row = pidx >> 7;
        int col = (pidx & 127) * 2;
        float2 v = *(const float2*)&agg[(((size_t)(col >> 6)) * NNODES + nb + row) * EMDIM + (col & 63)];
        v2bf b; b[0] = (__bf16)v.x; b[1] = (__bf16)v.y;
        *(v2bf*)&hc[row * 256 + col] = b;
    }
    // single wave: LDS ops are in-order within the wave

    const int half = lane >> 4, m = lane & 15;

    // mid = leaky(hcat @ Wf1 + bf1)    (16x256x64)
    {
        v16bf a[8];
#pragma unroll
        for (int kt = 0; kt < 8; ++kt)
            a[kt] = fragA_lds(hc, HEADS * EMDIM, m, half, kt * 32);
#pragma unroll
        for (int nt = 0; nt < 4; ++nt) {
            v8f acc = {};
#pragma unroll
            for (int kt = 0; kt < 8; ++kt)
                acc = wmma_bf16(a[kt], fragB_packed(Wf1p, nt * 8 + kt, lane), acc);
            int n = nt * 16 + m;
            storeC_lds(mid, HIDIM, n, half, acc, bf1[n], true);
        }
    }

    // out = mid @ Wf2 + bf2            (16x64x128)
    {
        v16bf a[2];
#pragma unroll
        for (int kt = 0; kt < 2; ++kt)
            a[kt] = fragA_lds(mid, HIDIM, m, half, kt * 32);
#pragma unroll
        for (int nt = 0; nt < 8; ++nt) {
            v8f acc = {};
#pragma unroll
            for (int kt = 0; kt < 2; ++kt)
                acc = wmma_bf16(a[kt], fragB_packed(Wf2p, nt * 2 + kt, lane), acc);
            int n = nt * 16 + m;
            float bias = bf2[n];
#pragma unroll
            for (int r = 0; r < 8; ++r)
                outs[(r + 8 * half) * OUTDIM + n] = acc[r] + bias;
        }
    }

    // coalesced dump (b128)
    float* g = out + (size_t)nb * OUTDIM;
#pragma unroll
    for (int i = 0; i < 16; ++i) {
        int o = i * 128 + lane * 4;
        *(float4*)&g[o] = *(const float4*)&outs[o];
    }
}

// ---------------------------------------------------------------------------
extern "C" void kernel_launch(void* const* d_in, const int* in_sizes, int n_in,
                              void* d_out, int out_size, void* d_ws, size_t ws_size,
                              hipStream_t stream) {
    const float* x   = (const float*)d_in[0];
    const int*   ei  = (const int*)  d_in[1];
    const float* Wi1 = (const float*)d_in[2];
    const float* bi1 = (const float*)d_in[3];
    const float* Wi2 = (const float*)d_in[4];
    const float* bi2 = (const float*)d_in[5];
    const float* Wa1 = (const float*)d_in[6];
    const float* ba1 = (const float*)d_in[7];
    const float* Wa2 = (const float*)d_in[8];
    const float* ba2 = (const float*)d_in[9];
    const float* Wf1 = (const float*)d_in[10];
    const float* bf1 = (const float*)d_in[11];
    const float* Wf2 = (const float*)d_in[12];
    const float* bf2 = (const float*)d_in[13];
    float* out = (float*)d_out;

    char* ws = (char*)d_ws;
    size_t off = 0;
    auto carve = [&](size_t bytes) {
        char* p = ws + off;
        off += (bytes + 255) & ~(size_t)255;
        return p;
    };
    __bf16* hs    = (__bf16*)carve((size_t)HEADS * NNODES * EMDIM * 2);  // 51.2 MB
    __bf16* ssrc  = (__bf16*)carve((size_t)HEADS * NNODES * HIDIM * 2);  // 51.2 MB
    __bf16* sdst  = (__bf16*)carve((size_t)HEADS * NNODES * HIDIM * 2);  // 51.2 MB
    float*  e_ws  = (float*) carve((size_t)HEADS * NEDGES * 4);          // 25.6 MB
    float*  esum  = (float*) carve((size_t)HEADS * NNODES * 4);          //  1.6 MB
    float*  agg   = (float*) carve((size_t)HEADS * NNODES * EMDIM * 4);  // 102.4 MB
    __bf16* Wi1p  = (__bf16*)carve((size_t)HEADS * INDIM * EMDIM * 2);
    __bf16* Wi2p  = (__bf16*)carve((size_t)HEADS * EMDIM * EMDIM * 2);
    __bf16* Wa1ap = (__bf16*)carve((size_t)HEADS * EMDIM * HIDIM * 2);
    __bf16* Wa1bp = (__bf16*)carve((size_t)HEADS * EMDIM * HIDIM * 2);
    __bf16* Wf1p  = (__bf16*)carve((size_t)(HEADS * EMDIM) * HIDIM * 2);
    __bf16* Wf2p  = (__bf16*)carve((size_t)HIDIM * OUTDIM * 2);

    auto pack = [&](const float* src, __bf16* dst, int K, int N, int batch,
                    int rowOff, int srcK) {
        int total = batch * K * N;
        pack_w_kernel<<<(total + 255) / 256, 256, 0, stream>>>(
            src, dst, K, N, batch, rowOff, (size_t)srcK * N);
    };
    pack(Wi1, Wi1p, INDIM, EMDIM, HEADS, 0, INDIM);
    pack(Wi2, Wi2p, EMDIM, EMDIM, HEADS, 0, EMDIM);
    pack(Wa1, Wa1ap, EMDIM, HIDIM, HEADS, 0, 2 * EMDIM);
    pack(Wa1, Wa1bp, EMDIM, HIDIM, HEADS, EMDIM, 2 * EMDIM);
    pack(Wf1, Wf1p, HEADS * EMDIM, HIDIM, 1, 0, HEADS * EMDIM);
    pack(Wf2, Wf2p, HIDIM, OUTDIM, 1, 0, HIDIM);

    hipMemsetAsync(esum, 0, (size_t)HEADS * NNODES * 4, stream);
    hipMemsetAsync(agg,  0, (size_t)HEADS * NNODES * EMDIM * 4, stream);

    node_embed_kernel<<<NNODES / 16, 128, 0, stream>>>(
        x, Wi1p, bi1, Wi2p, bi2, Wa1ap, Wa1bp, hs, ssrc, sdst);

    edge_attn_kernel<<<NEDGES / 8, 256, 0, stream>>>(
        ei, ssrc, sdst, ba1, Wa2, ba2, e_ws, esum);

    edge_aggr_kernel<<<NEDGES / 8, 256, 0, stream>>>(
        ei, hs, e_ws, esum, agg);

    final_mlp_kernel<<<NNODES / 16, 32, 0, stream>>>(
        agg, Wf1p, bf1, Wf2p, bf2, out);
}
